// Network_32195074851422
// MI455X (gfx1250) — matrix operand order repaired and checked by reading, not compiled, not run on hardware
//
#include <hip/hip_runtime.h>
#include <hip/hip_bf16.h>

typedef __attribute__((ext_vector_type(16))) _Float16 v16h;
typedef __attribute__((ext_vector_type(8)))  float    v8f;
typedef __attribute__((ext_vector_type(4)))  unsigned int v4u;

#define N_NODES 64
#define VPTS    8192
#define TM      16

// Output tuple layout (flat, in return order): c, d, ei, delta, mean, std
#define C_OFF     0        // [V,3]   -> 24576
#define D_OFF     24576    // [V,1]   -> 8192
#define EI_OFF    32768    // [64,32] -> 2048
#define DELTA_OFF 34816    // [64,3]  -> 192
#define MEAN_OFF  35008    // [64,8]  -> 512
#define STD_OFF   35520    // [64,8]  -> 512

// Packed B-fragment arena in d_ws (f16, 512 halves = 1 KB per fragment).
// Per node: L1 tiles0..3 x kc0..2 (12) | L2 (8) | L3 (8) | L4 tiles0..15 x kc0..1 (32)
#define FRAGS_PER_NODE 60
#define L1_FOFF 0
#define L2_FOFF 12
#define L3_FOFF 20
#define L4_FOFF 28
#define HEAD_FBASE (N_NODES * FRAGS_PER_NODE)   // 3840: cW1, 4 tiles x 8 kc
#define TOTAL_FRAGS (HEAD_FBASE + 32)           // 3872 frags * 1KB
#define EI16_HOFF  ((size_t)TOTAL_FRAGS << 9)   // f16 ei copy [64][32] after frags

union AFrag { v16h v; v4u q[2]; unsigned int u[8]; _Float16 h[16]; };

#define WMMA(accv, av, bv) \
    __builtin_amdgcn_wmma_f32_16x16x32_f16(false, (av), false, (bv), (short)0, (accv), false, false)

// Load an A fragment (16x32 f16 slice) from a row-major f16 LDS array.
// Halves 0..7 cover contiguous K=8g..8g+7, halves 8..15 cover K=16+8g..15+8g+8.
#define LOAD_AFRAG(dst, Hin, l16, g, k0)                                              \
    do {                                                                               \
        (dst).q[0] = *reinterpret_cast<const v4u*>(&(Hin)[(l16)][(k0) + 8 * (g)]);     \
        (dst).q[1] = *reinterpret_cast<const v4u*>(&(Hin)[(l16)][(k0) + 16 + 8 * (g)]);\
    } while (0)

// Load a packed B fragment: one 32-byte read per lane.
#define LOAD_BFRAG(dst, pk, fragIdx, lane)                                            \
    (dst).v = *reinterpret_cast<const v16h*>((pk) + ((size_t)(fragIdx) << 9) + ((lane) << 4))

// ---------------------------------------------------------------------------
// Kernel 0: pack f32 weights into f16 WMMA B-fragment layout (lane-major).
// ---------------------------------------------------------------------------
__global__ __launch_bounds__(32) void pack_weights_kernel(
    const float* __restrict__ fW1, const float* __restrict__ fW2,
    const float* __restrict__ fW3, const float* __restrict__ fW4,
    const float* __restrict__ cW1,
    _Float16* __restrict__ pk)
{
    const int f    = blockIdx.x;
    const int lane = threadIdx.x;
    const int g    = lane >> 4;
    const int nn   = lane & 15;

    const float* W; int OS, n0, k0, kmax;
    if (f < HEAD_FBASE) {
        const int node = f / FRAGS_PER_NODE;
        int r = f - node * FRAGS_PER_NODE;
        if (r < L2_FOFF) {                 // layer 1: [71,64], K padded to 96
            const int tile = r / 3, kc = r - tile * 3;
            W = fW1 + node * 71 * 64; OS = 64; n0 = tile * 16; k0 = kc * 32; kmax = 71;
        } else if (r < L3_FOFF) {          // layer 2: [64,64]
            r -= L2_FOFF; const int tile = r >> 1, kc = r & 1;
            W = fW2 + node * 64 * 64; OS = 64; n0 = tile * 16; k0 = kc * 32; kmax = 64;
        } else if (r < L4_FOFF) {          // layer 3: [64,64]
            r -= L3_FOFF; const int tile = r >> 1, kc = r & 1;
            W = fW3 + node * 64 * 64; OS = 64; n0 = tile * 16; k0 = kc * 32; kmax = 64;
        } else {                           // layer 4: [64,256]
            r -= L4_FOFF; const int tile = r >> 1, kc = r & 1;
            W = fW4 + node * 64 * 256; OS = 256; n0 = tile * 16; k0 = kc * 32; kmax = 64;
        }
    } else {                               // head cW1: [256,64]
        const int r = f - HEAD_FBASE;
        const int tile = r >> 3, kc = r & 7;
        W = cW1; OS = 64; n0 = tile * 16; k0 = kc * 32; kmax = 256;
    }

    AFrag b;
#pragma unroll
    for (int hh = 0; hh < 16; ++hh) {      // ISA B layout: K = k0 + 8g + hh (hh<8), k0 + 16 + 8g + hh-8 (hh>=8)
        const int r2 = hh >> 3, p = hh & 7;
        const int K  = k0 + 16 * r2 + 8 * g + p;
        b.h[hh] = (K < kmax) ? (_Float16)W[K * OS + n0 + nn] : (_Float16)0.f;
    }
    *reinterpret_cast<v16h*>(pk + ((size_t)f << 9) + (lane << 4)) = b.v;
}

// ---------------------------------------------------------------------------
// Kernel 1: encoder + decoder (tiny, B=1). One block per node, plain VALU.
// Also emits an f16 copy of ei into the fragment arena for broadcast A-frags.
// ---------------------------------------------------------------------------
__global__ __launch_bounds__(128) void encdec_kernel(
    const float* __restrict__ t_ped,  const float* __restrict__ pose,
    const float* __restrict__ ecW1,   const float* __restrict__ ecb1,
    const float* __restrict__ ecW21,  const float* __restrict__ ecb21,
    const float* __restrict__ ecW22,  const float* __restrict__ ecb22,
    const float* __restrict__ dcW1,   const float* __restrict__ dcb1,
    const float* __restrict__ dcW21,  const float* __restrict__ dcb21,
    const float* __restrict__ dcW22,  const float* __restrict__ dcb22,
    _Float16* __restrict__ pk,
    float* __restrict__ out)
{
    const int n   = blockIdx.x;
    const int tid = threadIdx.x;
    __shared__ float enc_in[88];
    __shared__ float h[64];
    __shared__ float dec_in[80];
    __shared__ float hd[64];

    if (tid < 16)            enc_in[tid] = t_ped[tid];
    else if (tid < 88)       enc_in[tid] = pose[tid - 16];
    __syncthreads();

    if (tid < 64) {
        float acc = ecb1[n * 64 + tid];
        for (int i = 0; i < 88; ++i)
            acc += enc_in[i] * ecW1[(n * 88 + i) * 64 + tid];
        h[tid] = acc > 0.f ? acc : 0.f;
    }
    __syncthreads();

    if (tid < 8) {
        float m = ecb21[n * 8 + tid];
        float s = ecb22[n * 8 + tid];
        for (int j = 0; j < 64; ++j) {
            m += h[j] * ecW21[(n * 64 + j) * 8 + tid];
            s += h[j] * ecW22[(n * 64 + j) * 8 + tid];
        }
        out[MEAN_OFF + n * 8 + tid] = m;
        out[STD_OFF  + n * 8 + tid] = 1.f / (1.f + expf(-s));
        dec_in[tid] = m;                       // z = mean (eval path)
    }
    if (tid >= 8 && tid < 80) dec_in[tid] = pose[tid - 8];
    __syncthreads();

    if (tid < 64) {
        float acc = dcb1[n * 64 + tid];
        for (int i = 0; i < 80; ++i)
            acc += dec_in[i] * dcW1[(n * 80 + i) * 64 + tid];
        hd[tid] = acc > 0.f ? acc : 0.f;
    }
    __syncthreads();

    if (tid < 32) {
        float acc = dcb21[n * 32 + tid];
        for (int j = 0; j < 64; ++j)
            acc += hd[j] * dcW21[(n * 64 + j) * 32 + tid];
        out[EI_OFF + n * 32 + tid] = acc;
        (pk + EI16_HOFF)[n * 32 + tid] = (_Float16)acc;
    }
    if (tid >= 32 && tid < 35) {
        const int o = tid - 32;
        float acc = dcb22[n * 3 + o];
        for (int j = 0; j < 64; ++j)
            acc += hd[j] * dcW22[(n * 64 + j) * 3 + o];
        out[DELTA_OFF + n * 3 + o] = acc;
    }
}

// ---------------------------------------------------------------------------
// Kernel 2: fused feature field + blend + NeRF head.
// One independent 32-thread wave per workgroup, 16 v-rows per wave, ALL
// channels computed by the wave (no inter-wave barriers: single-wave WG ->
// barrier ops are hardware NOPs). A fragments loaded once per k-chunk and
// reused across channel tiles; blend accumulator + wsum live in registers.
// ---------------------------------------------------------------------------
__global__ __launch_bounds__(32, 1) void feat_blend_head_kernel(
    const float* __restrict__ lce,   // [V,64,39]
    const float* __restrict__ bwg,   // [V,64]
    const _Float16* __restrict__ pk, // packed B fragments + ei16 (d_ws)
    const float* __restrict__ fb1, const float* __restrict__ fb2,
    const float* __restrict__ fb3, const float* __restrict__ fb4,
    const float* __restrict__ cb1,
    const float* __restrict__ cW2, const float* __restrict__ cb2,
    const float* __restrict__ dW1, const float* __restrict__ db1,
    float* __restrict__ out)
{
    __shared__ __align__(16) _Float16 Xs[TM][64];   // lce cols (39 real, pad 0)
    __shared__ __align__(16) _Float16 HsA[TM][64];
    __shared__ __align__(16) _Float16 HsB[TM][64];
    __shared__ __align__(16) float    fblend[TM][256];
    __shared__ __align__(16) _Float16 fb16[TM][256];
    __shared__ float    netc[TM][64];

    const int lane = threadIdx.x;
    const int l16  = lane & 15;
    const int g    = lane >> 4;
    const int v0   = blockIdx.x * TM;
    const _Float16* ei16 = pk + EI16_HOFF;

    // zero Xs once; cols >=40 are never rewritten and stay zero (K pad 71->96)
    {
        v4u z = {0u, 0u, 0u, 0u};
        for (int j = lane; j < (TM * 64) / 8; j += 32)
            *reinterpret_cast<v4u*>(&Xs[0][0] + j * 8) = z;
    }
    __syncthreads();

    float facc[16][8];                 // blended f: 16 ch-tiles x 8 rows/lane
    float wsumr[8];
#pragma unroll
    for (int t = 0; t < 16; ++t)
#pragma unroll
        for (int i = 0; i < 8; ++i) facc[t][i] = 0.f;
#pragma unroll
    for (int i = 0; i < 8; ++i) wsumr[i] = 0.f;

    for (int n = 0; n < N_NODES; ++n) {
        const int nfrag = n * FRAGS_PER_NODE;
        const _Float16* ein = ei16 + n * 32;
        __syncthreads();
        // ---- stage lce tile (16 rows x 39 cols, col 39 zeroed)
        for (int j = lane; j < TM * 20; j += 32) {
            const int row = j / 20, cp = j - row * 20;
            const float* lr = lce + ((size_t)(v0 + row) * N_NODES + n) * 39;
            const int c0 = cp * 2;
            const float f0 = lr[c0];
            const float f1 = (c0 + 1 < 39) ? lr[c0 + 1] : 0.f;
            Xs[row][c0]     = (_Float16)f0;
            Xs[row][c0 + 1] = (_Float16)f1;
        }
        float wloc[8];
#pragma unroll
        for (int i = 0; i < 8; ++i) {   // blend weights for this lane's 8 rows
            wloc[i] = bwg[(size_t)(v0 + 8 * g + i) * N_NODES + n] + 1e-6f;
            wsumr[i] += wloc[i];
        }
        if (lane == 0 && n + 1 < N_NODES)
            __builtin_prefetch(pk + ((size_t)(n + 1) * FRAGS_PER_NODE << 9), 0, 0);
        __syncthreads();

        // ---- layer 1: K=96 (kc0 = broadcast ei, kc1/kc2 from Xs), relu->HsA
        {
            v8f acc[4] = {};
            AFrag a, b;
            a.q[0] = *reinterpret_cast<const v4u*>(ein + 8 * g);
            a.q[1] = *reinterpret_cast<const v4u*>(ein + 16 + 8 * g);
#pragma unroll
            for (int t = 0; t < 4; ++t) {
                LOAD_BFRAG(b, pk, nfrag + L1_FOFF + t * 3 + 0, lane);
                acc[t] = WMMA(acc[t], a.v, b.v);
            }
            LOAD_AFRAG(a, Xs, l16, g, 0);
#pragma unroll
            for (int t = 0; t < 4; ++t) {
                LOAD_BFRAG(b, pk, nfrag + L1_FOFF + t * 3 + 1, lane);
                acc[t] = WMMA(acc[t], a.v, b.v);
            }
            LOAD_AFRAG(a, Xs, l16, g, 32);
#pragma unroll
            for (int t = 0; t < 4; ++t) {
                LOAD_BFRAG(b, pk, nfrag + L1_FOFF + t * 3 + 2, lane);
                acc[t] = WMMA(acc[t], a.v, b.v);
            }
#pragma unroll
            for (int t = 0; t < 4; ++t) {
                const float bias = fb1[n * 64 + t * 16 + l16];
#pragma unroll
                for (int i = 0; i < 8; ++i) {
                    float vv = acc[t][i] + bias;
                    HsA[i + 8 * g][t * 16 + l16] = (_Float16)(vv > 0.f ? vv : 0.f);
                }
            }
        }
        __syncthreads();

        // ---- layer 2: HsA -> HsB
        {
            v8f acc[4] = {};
            AFrag a, b;
            LOAD_AFRAG(a, HsA, l16, g, 0);
#pragma unroll
            for (int t = 0; t < 4; ++t) {
                LOAD_BFRAG(b, pk, nfrag + L2_FOFF + t * 2 + 0, lane);
                acc[t] = WMMA(acc[t], a.v, b.v);
            }
            LOAD_AFRAG(a, HsA, l16, g, 32);
#pragma unroll
            for (int t = 0; t < 4; ++t) {
                LOAD_BFRAG(b, pk, nfrag + L2_FOFF + t * 2 + 1, lane);
                acc[t] = WMMA(acc[t], a.v, b.v);
            }
#pragma unroll
            for (int t = 0; t < 4; ++t) {
                const float bias = fb2[n * 64 + t * 16 + l16];
#pragma unroll
                for (int i = 0; i < 8; ++i) {
                    float vv = acc[t][i] + bias;
                    HsB[i + 8 * g][t * 16 + l16] = (_Float16)(vv > 0.f ? vv : 0.f);
                }
            }
        }
        __syncthreads();

        // ---- layer 3: HsB -> HsA
        {
            v8f acc[4] = {};
            AFrag a, b;
            LOAD_AFRAG(a, HsB, l16, g, 0);
#pragma unroll
            for (int t = 0; t < 4; ++t) {
                LOAD_BFRAG(b, pk, nfrag + L3_FOFF + t * 2 + 0, lane);
                acc[t] = WMMA(acc[t], a.v, b.v);
            }
            LOAD_AFRAG(a, HsB, l16, g, 32);
#pragma unroll
            for (int t = 0; t < 4; ++t) {
                LOAD_BFRAG(b, pk, nfrag + L3_FOFF + t * 2 + 1, lane);
                acc[t] = WMMA(acc[t], a.v, b.v);
            }
#pragma unroll
            for (int t = 0; t < 4; ++t) {
                const float bias = fb3[n * 64 + t * 16 + l16];
#pragma unroll
                for (int i = 0; i < 8; ++i) {
                    float vv = acc[t][i] + bias;
                    HsA[i + 8 * g][t * 16 + l16] = (_Float16)(vv > 0.f ? vv : 0.f);
                }
            }
        }
        __syncthreads();

        // ---- layer 4 (no relu) + fused blend accumulate (A hoisted, 16 tiles)
        {
            AFrag a0, a1, b;
            LOAD_AFRAG(a0, HsA, l16, g, 0);
            LOAD_AFRAG(a1, HsA, l16, g, 32);
#pragma unroll
            for (int t = 0; t < 16; ++t) {
                v8f acc = {};
                LOAD_BFRAG(b, pk, nfrag + L4_FOFF + t * 2 + 0, lane);
                acc = WMMA(acc, a0.v, b.v);
                LOAD_BFRAG(b, pk, nfrag + L4_FOFF + t * 2 + 1, lane);
                acc = WMMA(acc, a1.v, b.v);
                const float bias = fb4[n * 256 + t * 16 + l16];
#pragma unroll
                for (int i = 0; i < 8; ++i)
                    facc[t][i] += wloc[i] * (acc[i] + bias);
            }
        }
    }

    __syncthreads();
    // ---- normalize blend, stage f32 + f16 copies in LDS
#pragma unroll
    for (int t = 0; t < 16; ++t) {
        const int ch = t * 16 + l16;
#pragma unroll
        for (int i = 0; i < 8; ++i) {
            const int row = i + 8 * g;
            const float vv = facc[t][i] / wsumr[i];
            fblend[row][ch] = vv;
            fb16[row][ch]   = (_Float16)vv;
        }
    }
    __syncthreads();

    // ---- NeRF head: netc = relu(fblend @ cW1 + cb1)  (16x256 @ 256x64)
    {
        v8f acc[4] = {};
#pragma unroll
        for (int kc = 0; kc < 8; ++kc) {
            AFrag a, b;
            LOAD_AFRAG(a, fb16, l16, g, kc * 32);
#pragma unroll
            for (int t = 0; t < 4; ++t) {
                LOAD_BFRAG(b, pk, HEAD_FBASE + t * 8 + kc, lane);
                acc[t] = WMMA(acc[t], a.v, b.v);
            }
        }
#pragma unroll
        for (int t = 0; t < 4; ++t) {
            const float bias = cb1[t * 16 + l16];
#pragma unroll
            for (int i = 0; i < 8; ++i) {
                float vv = acc[t][i] + bias;
                netc[i + 8 * g][t * 16 + l16] = vv > 0.f ? vv : 0.f;
            }
        }
    }
    __syncthreads();

    // ---- c = netc @ cW2 + cb2 ; d = relu(fblend @ dW1 + db1)
    for (int j = lane; j < TM * 3; j += 32) {
        const int row = j / 3, cidx = j - row * 3;
        float acc = cb2[cidx];
        for (int h2 = 0; h2 < 64; ++h2) acc += netc[row][h2] * cW2[h2 * 3 + cidx];
        out[C_OFF + (v0 + row) * 3 + cidx] = acc;
    }
    if (lane < TM) {
        float acc = db1[0];
        for (int ch = 0; ch < 256; ++ch) acc += fblend[lane][ch] * dW1[ch];
        out[D_OFF + v0 + lane] = acc > 0.f ? acc : 0.f;
    }
}

// ---------------------------------------------------------------------------
extern "C" void kernel_launch(void* const* d_in, const int* in_sizes, int n_in,
                              void* d_out, int out_size, void* d_ws, size_t ws_size,
                              hipStream_t stream) {
    const float* t_ped = (const float*)d_in[0];
    const float* pose  = (const float*)d_in[1];
    const float* lce   = (const float*)d_in[2];
    const float* bwg   = (const float*)d_in[3];
    const float* ecW1  = (const float*)d_in[4];
    const float* ecb1  = (const float*)d_in[5];
    const float* ecW21 = (const float*)d_in[6];
    const float* ecb21 = (const float*)d_in[7];
    const float* ecW22 = (const float*)d_in[8];
    const float* ecb22 = (const float*)d_in[9];
    const float* dcW1  = (const float*)d_in[10];
    const float* dcb1  = (const float*)d_in[11];
    const float* dcW21 = (const float*)d_in[12];
    const float* dcb21 = (const float*)d_in[13];
    const float* dcW22 = (const float*)d_in[14];
    const float* dcb22 = (const float*)d_in[15];
    const float* fW1   = (const float*)d_in[16];
    const float* fb1   = (const float*)d_in[17];
    const float* fW2   = (const float*)d_in[18];
    const float* fb2   = (const float*)d_in[19];
    const float* fW3   = (const float*)d_in[20];
    const float* fb3   = (const float*)d_in[21];
    const float* fW4   = (const float*)d_in[22];
    const float* fb4   = (const float*)d_in[23];
    const float* cW1   = (const float*)d_in[24];
    const float* cb1   = (const float*)d_in[25];
    const float* cW2   = (const float*)d_in[26];
    const float* cb2   = (const float*)d_in[27];
    const float* dW1   = (const float*)d_in[28];
    const float* db1   = (const float*)d_in[29];

    float*     out = (float*)d_out;
    _Float16*  pk  = (_Float16*)d_ws;

    pack_weights_kernel<<<TOTAL_FRAGS, 32, 0, stream>>>(fW1, fW2, fW3, fW4, cW1, pk);

    encdec_kernel<<<N_NODES, 128, 0, stream>>>(
        t_ped, pose, ecW1, ecb1, ecW21, ecb21, ecW22, ecb22,
        dcW1, dcb1, dcW21, dcb21, dcW22, dcb22, pk, out);

    feat_blend_head_kernel<<<VPTS / TM, 32, 0, stream>>>(
        lce, bwg, pk, fb1, fb2, fb3, fb4, cb1, cW2, cb2, dW1, db1, out);
}
